// iWiViG_44590350467899
// MI455X (gfx1250) — matrix operand order repaired and checked by reading, not compiled, not run on hardware
//
#include <hip/hip_runtime.h>
#include <hip/hip_bf16.h>
#include <math.h>

// ---------------- problem constants ----------------
#define Bq    16
#define Cch   192
#define Hh    56
#define Ww    56
#define Nn    (Hh * Ww)          // 3136
#define Kk    9
#define NCLS  100
#define NCLSP 112                // padded to 7 tiles of 16
#define PDIM  1024
#define NODES (Bq * Nn)          // 50176
#define EDGES (NODES * Kk)       // 451584

// ---------------- WMMA types (CDNA5 gfx1250) ----------------
typedef __attribute__((ext_vector_type(16))) __bf16 v16bf;
typedef __attribute__((ext_vector_type(8)))  float  v8f;
typedef __attribute__((ext_vector_type(4)))  unsigned int u32x4;
typedef __attribute__((ext_vector_type(8)))  unsigned int u32x8;

union FragBF { v16bf v; uint4 u[2]; };

__device__ __forceinline__ unsigned short f2bf(float f) {
    unsigned int u = __float_as_uint(f);
    unsigned int r = u + 0x7FFFu + ((u >> 16) & 1u);   // round-to-nearest-even
    return (unsigned short)(r >> 16);
}

// =====================================================================
// prep_weights: bf16 conv weights (padded), bf16 transposed gnn weights,
//               zero the pooled-feature accumulator g.
// =====================================================================
__global__ __launch_bounds__(256)
void prep_weights_kernel(const float* __restrict__ conv_w,
                         const float* __restrict__ gnn_w,
                         unsigned short* __restrict__ convb,
                         unsigned short* __restrict__ gnnwt,
                         float* __restrict__ g) {
    int t = blockIdx.x * 256 + threadIdx.x;
    if (t < NCLSP * Cch) {
        int cls = t / Cch, ci = t % Cch;
        convb[t] = (cls < NCLS) ? f2bf(conv_w[cls * Cch + ci]) : (unsigned short)0;
    }
    if (t < Cch * Cch) {
        int nc = t / Cch, k = t % Cch;          // gnnwt[nc*Cch + k] = gnn_w[k][nc]
        gnnwt[t] = f2bf(gnn_w[k * Cch + nc]);
    }
    if (t < Bq * Cch) g[t] = 0.0f;
}

// =====================================================================
// prep_nodes: [B,C,H,W] -> [node,C]; emit raw bf16 and L2-normalized bf16.
// =====================================================================
__global__ __launch_bounds__(256)
void prep_nodes_kernel(const float* __restrict__ imf,
                       unsigned short* __restrict__ rawb,
                       unsigned short* __restrict__ xnb) {
    __shared__ float red[256];
    const int node = blockIdx.x;
    const int b = node / Nn, n = node % Nn;
    const int c = threadIdx.x;
    float v = 0.0f;
    if (c < Cch) {
        v = imf[((size_t)b * Cch + c) * Nn + n];
        rawb[(size_t)node * Cch + c] = f2bf(v);
    }
    red[c] = v * v;
    __syncthreads();
    for (int off = 128; off > 0; off >>= 1) {
        if (c < off) red[c] += red[c + off];
        __syncthreads();
    }
    const float inv = 1.0f / fmaxf(sqrtf(red[0]), 1e-12f);
    if (c < Cch) xnb[(size_t)node * Cch + c] = f2bf(v * inv);
}

// =====================================================================
// conv_gemm: patch_pred = raw_nodes[50176,192] @ conv_w^T  (bf16 WMMA)
// all 6 B fragments preloaded (one load clause, one wait, 6 WMMAs).
// =====================================================================
__global__ __launch_bounds__(32)
void conv_gemm_kernel(const unsigned short* __restrict__ xraw,
                      const unsigned short* __restrict__ wconv,
                      const float* __restrict__ conv_b,
                      float* __restrict__ patch) {
    const int r0 = blockIdx.x * 16;
    const int lane = threadIdx.x, l15 = lane & 15, kh = lane >> 4;
    const int bimg = r0 / Nn, n0 = r0 % Nn;     // 16-row tile never crosses image
    float* patchBase = patch + (size_t)bimg * NCLS * Nn + n0;

    FragBF A[6];
    const uint4* arow = (const uint4*)(xraw + (size_t)(r0 + l15) * Cch);
#pragma unroll
    for (int s = 0; s < 6; ++s) {               // A 16x32: K = 16*(i/8) + 8*kh + i%8
        A[s].u[0] = arow[s * 4 + kh];
        A[s].u[1] = arow[s * 4 + 2 + kh];
    }
    for (int ct = 0; ct < NCLSP / 16; ++ct) {
        const uint4* bcol = (const uint4*)(wconv + (size_t)(ct * 16 + l15) * Cch);
        FragBF Bf[6];
#pragma unroll
        for (int s = 0; s < 6; ++s) {           // B 32x16: K = 16*(i/8) + 8*kh + i%8
            Bf[s].u[0] = bcol[s * 4 + kh];
            Bf[s].u[1] = bcol[s * 4 + 2 + kh];
        }
        v8f acc = {};
#pragma unroll
        for (int s = 0; s < 6; ++s)
            acc = __builtin_amdgcn_wmma_f32_16x16x32_bf16(
                false, A[s].v, false, Bf[s].v, (short)0, acc, false, false);

        const int cls = ct * 16 + l15;
        if (cls < NCLS) {
            const float bb = conv_b[cls];
#pragma unroll
            for (int r = 0; r < 8; ++r)
                patchBase[(size_t)cls * Nn + (r + 8 * kh)] = acc[r] + bb;
        }
    }
}

// =====================================================================
// sim_topk: the hot kernel (30 GMAC). 2 waves / block, 64 rows / block.
// Column tiles (16 nodes x 192ch bf16 = 6 KB) are staged into LDS by the
// Tensor Data Mover (tensor_load_to_lds, TENSORcnt), double-buffered so
// the DMA for tile t+1 overlaps the 12 WMMAs of tile t. Each wave holds
// M=32 rows (two accumulators), so every staged tile feeds 4 WMMA sets.
// Running top-9 kept in registers of all 32 lanes.
// =====================================================================
__global__ __launch_bounds__(64)
void sim_topk_kernel(const unsigned short* __restrict__ xb,
                     int* __restrict__ nbr) {
    __shared__ __align__(16) unsigned short Btile[2][16 * Cch]; // 2 x 6144 B
    __shared__ float simt[2][32][17];

    const int img  = blockIdx.x / (Nn / 64);
    const int r0   = (blockIdx.x % (Nn / 64)) * 64;
    const int w    = threadIdx.x >> 5;          // wave id 0..1
    const int lane = threadIdx.x & 31, l15 = lane & 15, kh = lane >> 4;
    const int rbase = r0 + w * 32;              // this wave's 32 rows (local)

    // ---- preload A fragments: 2 row-halves x 6 k-steps (96 VGPRs) ----
    FragBF A0[6], A1[6];
    {
        const uint4* a0 = (const uint4*)(xb + (size_t)(img * Nn + rbase + l15) * Cch);
        const uint4* a1 = (const uint4*)(xb + (size_t)(img * Nn + rbase + 16 + l15) * Cch);
#pragma unroll
        for (int s = 0; s < 6; ++s) {
            A0[s].u[0] = a0[s * 4 + kh];     A0[s].u[1] = a0[s * 4 + 2 + kh];
            A1[s].u[0] = a1[s * 4 + kh];     A1[s].u[1] = a1[s * 4 + 2 + kh];
        }
    }

    const unsigned lds0 = (unsigned)(size_t)&Btile[0][0];
    const unsigned lds1 = (unsigned)(size_t)&Btile[1][0];

    // ---- TDM issue: 1-D copy of 6144 B (768 qwords) global -> LDS ----
    auto issueTdm = [&](int t, int buf) {
        unsigned long long ga =
            (unsigned long long)(size_t)(xb + (size_t)(img * Nn + t * 16) * Cch);
        u32x4 g0;
        g0[0] = 1u;                                            // count=1
        g0[1] = buf ? lds1 : lds0;                             // lds_addr
        g0[2] = (unsigned)ga;                                  // global_addr lo
        g0[3] = ((unsigned)(ga >> 32) & 0x01FFFFFFu) | 0x80000000u; // hi | type=2
        u32x8 g1;
        g1[0] = (3u << 16);          // data_size = 8 bytes
        g1[1] = (768u & 0xFFFFu) << 16;   // tensor_dim0 lo16 (=768)
        g1[2] = (1u << 16);          // tensor_dim0 hi | tensor_dim1 = 1
        g1[3] = (768u << 16);        // tensor_dim1 hi | tile_dim0 = 768
        g1[4] = 1u;                  // tile_dim1 = 1
        g1[5] = 0u; g1[6] = 0u; g1[7] = 0u;
        asm volatile("tensor_load_to_lds %0, %1" :: "s"(g0), "s"(g1) : "memory");
    };

    float best[Kk]; int bidx[Kk];
#pragma unroll
    for (int k = 0; k < Kk; ++k) { best[k] = -1e30f; bidx[k] = 0; }
    const int myrow = rbase + lane;             // row owned by this lane (local)

    // prologue: stage tile 0
    if (w == 0) {
        issueTdm(0, 0);
        __builtin_amdgcn_s_wait_tensorcnt(0);
    }
    __syncthreads();

    for (int t = 0; t < Nn / 16; ++t) {
        const int cur = t & 1;
        if (w == 0 && t + 1 < Nn / 16) issueTdm(t + 1, cur ^ 1);   // overlap DMA

        // B fragments from LDS (12 ds_load_b128, then 12 WMMAs)
        const uint4* bt = (const uint4*)&Btile[cur][0] + (size_t)l15 * 24;
        FragBF Bf[6];
#pragma unroll
        for (int s = 0; s < 6; ++s) {
            Bf[s].u[0] = bt[s * 4 + kh];
            Bf[s].u[1] = bt[s * 4 + 2 + kh];
        }
        v8f a0 = {}, a1 = {};
#pragma unroll
        for (int s = 0; s < 6; ++s) {
            a0 = __builtin_amdgcn_wmma_f32_16x16x32_bf16(
                false, A0[s].v, false, Bf[s].v, (short)0, a0, false, false);
            a1 = __builtin_amdgcn_wmma_f32_16x16x32_bf16(
                false, A1[s].v, false, Bf[s].v, (short)0, a1, false, false);
        }
#pragma unroll
        for (int r = 0; r < 8; ++r) {
            simt[w][r + 8 * kh][l15]      = a0[r];
            simt[w][16 + r + 8 * kh][l15] = a1[r];
        }
        __syncthreads();

        // top-9 scan: every lane owns one of the wave's 32 rows
        for (int c = 0; c < 16; ++c) {
            const int j = t * 16 + c;
            float v = simt[w][lane][c];
            if (j == myrow) v = -1e30f;         // exclude self (== sim - 4 trick)
            if (v > best[Kk - 1]) {             // strict > keeps index-stable ties
                best[Kk - 1] = v; bidx[Kk - 1] = j;
#pragma unroll
                for (int p = Kk - 1; p > 0; --p) {
                    if (best[p] > best[p - 1]) {
                        float tv = best[p]; best[p] = best[p - 1]; best[p - 1] = tv;
                        int ti = bidx[p]; bidx[p] = bidx[p - 1]; bidx[p - 1] = ti;
                    }
                }
            }
        }
        if (w == 0) __builtin_amdgcn_s_wait_tensorcnt(0);  // next buffer ready
        __syncthreads();
    }

    const int node = img * Nn + myrow;
#pragma unroll
    for (int k = 0; k < Kk; ++k)
        nbr[node * Kk + k] = img * Nn + bidx[k];           // global src index
}

// =====================================================================
// gnn_gemm: h = relu(nodes @ gnn_w + b)   (bf16 WMMA, f32 out)
// =====================================================================
__global__ __launch_bounds__(32)
void gnn_gemm_kernel(const unsigned short* __restrict__ xb,
                     const unsigned short* __restrict__ wt,
                     const float* __restrict__ bias,
                     float* __restrict__ h) {
    const int r0 = blockIdx.x * 16;
    const int lane = threadIdx.x, l15 = lane & 15, kh = lane >> 4;

    FragBF A[6];
    const uint4* arow = (const uint4*)(xb + (size_t)(r0 + l15) * Cch);
#pragma unroll
    for (int s = 0; s < 6; ++s) {
        A[s].u[0] = arow[s * 4 + kh];
        A[s].u[1] = arow[s * 4 + 2 + kh];
    }
    for (int ct = 0; ct < Cch / 16; ++ct) {
        const uint4* bcol = (const uint4*)(wt + (size_t)(ct * 16 + l15) * Cch);
        FragBF Bf[6];
#pragma unroll
        for (int s = 0; s < 6; ++s) {
            Bf[s].u[0] = bcol[s * 4 + kh];
            Bf[s].u[1] = bcol[s * 4 + 2 + kh];
        }
        v8f acc = {};
#pragma unroll
        for (int s = 0; s < 6; ++s)
            acc = __builtin_amdgcn_wmma_f32_16x16x32_bf16(
                false, A[s].v, false, Bf[s].v, (short)0, acc, false, false);

        const int ncol = ct * 16 + l15;
        const float bb = bias[ncol];
#pragma unroll
        for (int r = 0; r < 8; ++r) {
            const float v = acc[r] + bb;
            h[(size_t)(r0 + r + 8 * kh) * Cch + ncol] = v > 0.0f ? v : 0.0f;
        }
    }
}

// =====================================================================
// edge_att: sigmoid(dot(h[src], h[dst])) — one wave per edge.
// =====================================================================
__global__ __launch_bounds__(256)
void edge_att_kernel(const float* __restrict__ h,
                     const int* __restrict__ nbr,
                     float* __restrict__ out) {
    const int wv = threadIdx.x >> 5, lane = threadIdx.x & 31;
    const int e = blockIdx.x * 8 + wv;          // EDGES % 8 == 0
    const int dst = e / Kk;
    const int src = nbr[e];
    const float* hs = h + (size_t)src * Cch;
    const float* hd = h + (size_t)dst * Cch;
    float s = 0.0f;
#pragma unroll
    for (int i = 0; i < Cch / 32; ++i)
        s += hs[lane + i * 32] * hd[lane + i * 32];
#pragma unroll
    for (int off = 16; off > 0; off >>= 1)
        s += __shfl_xor(s, off, 32);
    if (lane == 0) out[e] = 1.0f / (1.0f + expf(-s));
}

// =====================================================================
// aggregate: h' = h + mean_k h[nbr]; pool sum per image -> g[B,C].
// =====================================================================
__global__ __launch_bounds__(192)
void aggregate_kernel(const float* __restrict__ h,
                      const int* __restrict__ nbr,
                      float* __restrict__ g) {
    const int c = threadIdx.x;
    const int n0 = blockIdx.x * 64;
    const int b = n0 / Nn;
    float acc = 0.0f;
    for (int i = 0; i < 64; ++i) {
        const int n = n0 + i;
        float m = 0.0f;
#pragma unroll
        for (int k = 0; k < Kk; ++k) {
            const int s = nbr[n * Kk + k];
            m += h[(size_t)s * Cch + c];
        }
        acc += h[(size_t)n * Cch + c] + m * (1.0f / 9.0f);
    }
    atomicAdd(&g[b * Cch + c], acc);
}

// =====================================================================
// mlp_z: z = g@w1+b1 -> batch BN (train stats) -> exact GELU -> zact.
// =====================================================================
__global__ __launch_bounds__(1024)
void mlp_z_kernel(const float* __restrict__ g,
                  const float* __restrict__ w1,
                  const float* __restrict__ b1,
                  const float* __restrict__ gamma,
                  const float* __restrict__ beta,
                  float* __restrict__ zact) {
    const int j = threadIdx.x;
    float z[Bq];
#pragma unroll
    for (int i = 0; i < Bq; ++i) {
        float a = b1[j];
        for (int c = 0; c < Cch; ++c)
            a += g[i * Cch + c] * w1[(size_t)c * PDIM + j];
        z[i] = a;
    }
    float mu = 0.0f;
#pragma unroll
    for (int i = 0; i < Bq; ++i) mu += z[i];
    mu *= (1.0f / Bq);
    float s2 = 0.0f;
#pragma unroll
    for (int i = 0; i < Bq; ++i) { float d = z[i] - mu; s2 += d * d; }
    s2 *= (1.0f / Bq);                          // population variance (jnp.var)
    const float inv = rsqrtf(s2 + 1e-5f);
    const float ga = gamma[j], be = beta[j];
#pragma unroll
    for (int i = 0; i < Bq; ++i) {
        const float zn = (z[i] - mu) * inv * ga + be;
        zact[i * PDIM + j] = 0.5f * zn * (1.0f + erff(zn * 0.70710678f));
    }
}

// =====================================================================
// pred: pred = zact @ w2 + b2   (16x1024x100, trivially small)
// =====================================================================
__global__ __launch_bounds__(256)
void pred_gemm_kernel(const float* __restrict__ zact,
                      const float* __restrict__ w2,
                      const float* __restrict__ b2,
                      float* __restrict__ pred) {
    const int t = blockIdx.x * 256 + threadIdx.x;
    if (t >= Bq * NCLS) return;
    const int b = t / NCLS, k = t % NCLS;
    float a = b2[k];
    for (int jj = 0; jj < PDIM; ++jj)
        a += zact[b * PDIM + jj] * w2[(size_t)jj * NCLS + k];
    pred[t] = a;
}

// =====================================================================
static inline size_t alignUp(size_t x) { return (x + 255) & ~(size_t)255; }

extern "C" void kernel_launch(void* const* d_in, const int* in_sizes, int n_in,
                              void* d_out, int out_size, void* d_ws, size_t ws_size,
                              hipStream_t stream) {
    const float* imf    = (const float*)d_in[0];
    const float* conv_w = (const float*)d_in[1];
    const float* conv_b = (const float*)d_in[2];
    const float* gnn_w  = (const float*)d_in[3];
    const float* gnn_b  = (const float*)d_in[4];
    const float* w1     = (const float*)d_in[5];
    const float* b1     = (const float*)d_in[6];
    const float* bn_g   = (const float*)d_in[7];
    const float* bn_b   = (const float*)d_in[8];
    const float* w2     = (const float*)d_in[9];
    const float* b2     = (const float*)d_in[10];

    // d_out layout: pred[1600] | edge_att[451584] | patch_pred[5017600]
    float* out_pred = (float*)d_out;
    float* out_edge = out_pred + Bq * NCLS;
    float* out_patch = out_edge + EDGES;

    // workspace carve (all 256B aligned)
    char* w = (char*)d_ws;
    unsigned short* rawb = (unsigned short*)w;  w += alignUp((size_t)NODES * Cch * 2);
    unsigned short* xnb  = (unsigned short*)w;  w += alignUp((size_t)NODES * Cch * 2);
    float*          h    = (float*)w;           w += alignUp((size_t)NODES * Cch * 4);
    int*            nbr  = (int*)w;             w += alignUp((size_t)EDGES * 4);
    float*          g    = (float*)w;           w += alignUp((size_t)Bq * Cch * 4);
    float*          zact = (float*)w;           w += alignUp((size_t)Bq * PDIM * 4);
    unsigned short* cwb  = (unsigned short*)w;  w += alignUp((size_t)NCLSP * Cch * 2);
    unsigned short* gwt  = (unsigned short*)w;  w += alignUp((size_t)Cch * Cch * 2);

    prep_weights_kernel<<<(Cch * Cch + 255) / 256, 256, 0, stream>>>(
        conv_w, gnn_w, cwb, gwt, g);

    prep_nodes_kernel<<<NODES, 256, 0, stream>>>(imf, rawb, xnb);

    conv_gemm_kernel<<<NODES / 16, 32, 0, stream>>>(rawb, cwb, conv_b, out_patch);

    sim_topk_kernel<<<Bq * (Nn / 64), 64, 0, stream>>>(xnb, nbr);

    gnn_gemm_kernel<<<NODES / 16, 32, 0, stream>>>(xnb, gwt, gnn_b, h);

    edge_att_kernel<<<EDGES / 8, 256, 0, stream>>>(h, nbr, out_edge);

    aggregate_kernel<<<NODES / 64, 192, 0, stream>>>(h, nbr, g);

    mlp_z_kernel<<<1, 1024, 0, stream>>>(g, w1, b1, bn_g, bn_b, zact);

    pred_gemm_kernel<<<(Bq * NCLS + 255) / 256, 256, 0, stream>>>(zact, w2, b2, out_pred);
}
